// PAM_Module_9156870275087
// MI455X (gfx1250) — matrix operand order repaired and checked by reading, not compiled, not run on hardware
//
#include <hip/hip_runtime.h>
#include <cstdint>

// ---------------------------------------------------------------------------
// PAM module (DANet position attention) for MI455X / gfx1250.
// B=4, C=64, H=W=64, N=4096.
// Big GEMMs (f^T f and f @ attn^T) via v_wmma_f32_16x16x32_bf16, flash-style
// streaming softmax (256MB energy matrix never materialized), and TDM
// (tensor_load_to_lds) double-buffered tile staging overlapped with compute.
// ---------------------------------------------------------------------------

typedef __attribute__((ext_vector_type(16))) __bf16          v16bf;
typedef __attribute__((ext_vector_type(16))) unsigned short  v16us;
typedef __attribute__((ext_vector_type(8)))  float           v8f;
typedef __attribute__((ext_vector_type(4)))  unsigned int    v4ui;
typedef __attribute__((ext_vector_type(8)))  int             v8si;
typedef __attribute__((ext_vector_type(4)))  int             v4si;

#define Bb 4
#define Cc 64
#define Nn 4096

__device__ __forceinline__ unsigned short f2bf(float x) {
  unsigned int u = __builtin_bit_cast(unsigned int, x);
  unsigned int r = (u + 0x7fffu + ((u >> 16) & 1u)) >> 16;   // RNE
  return (unsigned short)r;
}

// A-matrix fragment (16x32 bf16), src layout [row][k], k contiguous.
// ISA 16-bit A layout: lane = M (0..15 twice), VGPR v holds K pair:
//   v<4 : K = 2v   (+8 for lanes 16..31);  v>=4: K = 2v+8 (+8 for upper lanes)
__device__ __forceinline__ v16bf gatherA(const unsigned short* src, int rowOff,
                                         int kOff, int stride, int lane) {
  const int hf = lane >> 4;
  const int m  = lane & 15;
  v16us t;
#pragma unroll
  for (int v = 0; v < 8; ++v) {
    int k = 2 * v + ((v >= 4) ? 8 : 0) + 8 * hf + kOff;
    const unsigned short* p = src + (rowOff + m) * stride + k;
    t[2 * v]     = p[0];
    t[2 * v + 1] = p[1];
  }
  return __builtin_bit_cast(v16bf, t);
}

// B-matrix fragment (32x16 bf16), src layout [n][k], k contiguous.
// B layout matching gatherA: lane = N (0..15 twice), VGPR v holds K pair:
//   v<4 : K = 2v   (+8 for lanes 16..31);  v>=4: K = 2v+8 (+8 for upper lanes)
__device__ __forceinline__ v16bf gatherB(const unsigned short* src, int nOff,
                                         int kOff, int stride, int lane) {
  const int hf = lane >> 4;
  const int n  = lane & 15;
  v16us t;
#pragma unroll
  for (int v = 0; v < 8; ++v) {
    int k = 2 * v + ((v >= 4) ? 8 : 0) + 8 * hf + kOff;
    const unsigned short* p = src + (nOff + n) * stride + k;
    t[2 * v]     = p[0];
    t[2 * v + 1] = p[1];
  }
  return __builtin_bit_cast(v16bf, t);
}

__device__ __forceinline__ v8f wmma_bf16(v16bf a, v16bf b, v8f c) {
  return __builtin_amdgcn_wmma_f32_16x16x32_bf16(
      false, a, false, b, (short)0, c, false, false);
}

// ---------------------------------------------------------------------------
// TDM: DMA a 16-row x 64-col bf16 sub-tile (rows strided by N elements in
// global memory) into LDS, contiguous [row][64] layout. 2D descriptor per
// CDNA5 ISA section 8 (groups 2/3 NULL-equivalent zeros for a 2D tensor).
// Tracked with TENSORcnt. 6-arg builtin form (clang-23 / therock headers).
// ---------------------------------------------------------------------------
struct U4s { unsigned int x[4]; };
struct U8s { unsigned int x[8]; };

__device__ __forceinline__ void tdm_load_tile16x64(const unsigned short* gsrc,
                                                   unsigned int lds_byte_addr) {
  unsigned long long va = (unsigned long long)(uintptr_t)gsrc;

  U4s g0s;
  g0s.x[0] = 1u;                                   // count=1, user descriptor
  g0s.x[1] = lds_byte_addr;                        // lds_addr
  g0s.x[2] = (unsigned int)va;                     // global_addr[31:0]
  g0s.x[3] = (unsigned int)((va >> 32) & 0x1FFFFFFu) | (2u << 30);  // type=2

  U8s g1s;
  g1s.x[0] = (1u << 16);                           // wg_mask=0, data_size=2B
  g1s.x[1] = (Nn & 0xFFFFu) << 16;                 // tensor_dim0 lo16 = 4096
  g1s.x[2] = ((Nn >> 16) & 0xFFFFu) | (16u << 16); // dim0 hi | tensor_dim1=16
  g1s.x[3] = (64u << 16);                          // dim1 hi=0 | tile_dim0=64
  g1s.x[4] = 16u;                                  // tile_dim1=16, tile_dim2=0
  g1s.x[5] = (unsigned int)Nn;                     // tensor_dim0_stride=4096
  g1s.x[6] = 0u;                                   // stride hi / dim1_stride
  g1s.x[7] = 0u;

  v4ui g0 = __builtin_bit_cast(v4ui, g0s);
  v8si g1 = __builtin_bit_cast(v8si, g1s);
  v4si z4 = (v4si){0, 0, 0, 0};
  v8si z8 = (v8si){0, 0, 0, 0, 0, 0, 0, 0};
  __builtin_amdgcn_tensor_load_to_lds(g0, g1, z4, z4, z8, 0);
}

// ---------------------------------------------------------------------------
// Kernel 1: conv3x3 (pad 1) + bias -> y, and per-channel sum/sumsq via atomics
// ---------------------------------------------------------------------------
__global__ __launch_bounds__(256) void conv_bn_stats_kernel(
    const float* __restrict__ x, const float* __restrict__ w,
    const float* __restrict__ bias, float* __restrict__ y,
    float* __restrict__ stats) {
  const int t  = blockIdx.x * 256 + threadIdx.x;   // [0, 4*64*4096)
  const int b  = t >> 18;
  const int r  = t & 262143;
  const int co = r >> 12;
  const int p  = r & 4095;
  const int h  = p >> 6;
  const int wq = p & 63;

  float acc = bias[co];
  const float* wb = w + co * (Cc * 9);
  const float* xb = x + (b << 18);
  for (int ci = 0; ci < Cc; ++ci) {
    const float* xc = xb + (ci << 12);
    const float* wk = wb + ci * 9;
#pragma unroll
    for (int dh = -1; dh <= 1; ++dh) {
      int hh = h + dh;
      if (hh < 0 || hh > 63) continue;
#pragma unroll
      for (int dw = -1; dw <= 1; ++dw) {
        int ww = wq + dw;
        if (ww < 0 || ww > 63) continue;
        acc += wk[(dh + 1) * 3 + (dw + 1)] * xc[(hh << 6) + ww];
      }
    }
  }
  y[t] = acc;

  __shared__ float rs[256], rq[256];
  rs[threadIdx.x] = acc;
  rq[threadIdx.x] = acc * acc;
  __syncthreads();
  for (int s = 128; s > 0; s >>= 1) {
    if (threadIdx.x < s) {
      rs[threadIdx.x] += rs[threadIdx.x + s];
      rq[threadIdx.x] += rq[threadIdx.x + s];
    }
    __syncthreads();
  }
  if (threadIdx.x == 0) {
    atomicAdd(&stats[co], rs[0]);
    atomicAdd(&stats[Cc + co], rq[0]);
  }
}

// ---------------------------------------------------------------------------
// Kernel 2: fold BN into per-channel affine  y' = a*y + bb
// ---------------------------------------------------------------------------
__global__ __launch_bounds__(64) void bn_finalize_kernel(
    const float* __restrict__ stats, const float* __restrict__ bn_scale,
    const float* __restrict__ bn_bias, float* __restrict__ coef) {
  const int c = threadIdx.x;
  const float inv_n = 1.0f / (float)(Bb * Nn);
  float mean = stats[c] * inv_n;
  float var  = stats[Cc + c] * inv_n - mean * mean;
  float a    = bn_scale[c] * rsqrtf(var + 1e-5f);
  coef[c]      = a;
  coef[Cc + c] = bn_bias[c] - mean * a;
}

// ---------------------------------------------------------------------------
// Kernel 3: apply affine + PReLU; feat (f32, in place) + bf16 copy for WMMA
// ---------------------------------------------------------------------------
__global__ __launch_bounds__(256) void bn_apply_prelu_kernel(
    float* __restrict__ y, unsigned short* __restrict__ fbf,
    const float* __restrict__ coef, const float* __restrict__ prelu_a) {
  const int t = blockIdx.x * 256 + threadIdx.x;
  const int c = (t >> 12) & 63;
  const float slope = prelu_a[0];
  float v = coef[c] * y[t] + coef[Cc + c];
  float f = (v >= 0.0f) ? v : slope * v;
  y[t]   = f;
  fbf[t] = f2bf(f);
}

// ---------------------------------------------------------------------------
// Kernel 4: flash-style self-attention with TDM double-buffered tiles.
// grid = (N/64, B), block = 128 (4 waves). Wave w owns rows i0 + 16w .. +16,
// and DMAs channel rows [16w, 16w+16) of each f tile.
// ---------------------------------------------------------------------------
__global__ __launch_bounds__(128) void pam_attention_kernel(
    const unsigned short* __restrict__ fbf, const float* __restrict__ feat,
    const float* __restrict__ gamma_p, float* __restrict__ out) {
  const int b    = blockIdx.y;
  const int i0   = blockIdx.x * 64;
  const int tid  = threadIdx.x;
  const int wv   = tid >> 5;
  const int lane = tid & 31;
  const int hf   = lane >> 4;
  const int n16  = lane & 15;

  __shared__ unsigned short fA[64 * 64];          // [i][c]  (A1: m=i, k=c)
  __shared__ unsigned short fBcj[2][64 * 64];     // [c][j]  (B2) - TDM target
  __shared__ unsigned short fBt[64 * 64];         // [j][c]  (B1) - transposed
  __shared__ unsigned short pls[4][16 * 64];      // per-wave P: [i][j] (A2)

  const unsigned short* fb = fbf + (b << 18);     // batch base, [c][N]

  // Issue TDM for tile 0 (each wave DMAs its 16 channel rows).
  {
    const unsigned short* gsrc = fb + (size_t)(wv * 16) * Nn;  // j0 = 0
    unsigned int lds_addr =
        (unsigned int)(uintptr_t)(void*)&fBcj[0][wv * 16 * 64];
    tdm_load_tile16x64(gsrc, lds_addr);
  }

  // stage f[:, i0:i0+64] as [i][c] (transposed; plain loads)
  for (int idx = tid; idx < 4096; idx += 128) {
    int i = idx & 63, c = idx >> 6;
    fA[i * 64 + c] = fb[(c << 12) + i0 + i];
  }
  __syncthreads();

  // A1 fragments for this wave's 16-row strip (K = c, 0..63)
  v16bf a1_0 = gatherA(fA, wv * 16, 0, 64, lane);
  v16bf a1_1 = gatherA(fA, wv * 16, 32, 64, lane);

  float m_run[8], l_run[8];
  v8f   o[4];
#pragma unroll
  for (int v = 0; v < 8; ++v) { m_run[v] = -1e30f; l_run[v] = 0.0f; }
#pragma unroll
  for (int cb = 0; cb < 4; ++cb) o[cb] = (v8f){0, 0, 0, 0, 0, 0, 0, 0};

  for (int n = 0; n < Nn / 64; ++n) {
    const int cur = n & 1;

    // Prefetch tile n+1 into the other buffer, then wait so tile n is
    // resident while tile n+1 streams in behind the compute below.
    if (n + 1 < Nn / 64) {
      const unsigned short* gsrc =
          fb + (size_t)(wv * 16) * Nn + (n + 1) * 64;
      unsigned int lds_addr =
          (unsigned int)(uintptr_t)(void*)&fBcj[cur ^ 1][wv * 16 * 64];
      tdm_load_tile16x64(gsrc, lds_addr);
      __builtin_amdgcn_s_wait_tensorcnt(1);
    } else {
      __builtin_amdgcn_s_wait_tensorcnt(0);
    }
    __syncthreads();   // fBcj[cur] visible to all waves

    // LDS->LDS transpose: fBt[j][c] = fBcj[cur][c][j]
    for (int idx = tid; idx < 4096; idx += 128) {
      int c = idx >> 6, j = idx & 63;
      fBt[j * 64 + c] = fBcj[cur][c * 64 + j];
    }
    __syncthreads();

    // energy tile E (16 rows x 64 cols), 4 sub-tiles of 16 cols
    v8f e[4];
#pragma unroll
    for (int jt = 0; jt < 4; ++jt) {
      v16bf b0 = gatherB(fBt, jt * 16, 0, 64, lane);
      v16bf b1 = gatherB(fBt, jt * 16, 32, 64, lane);
      v8f acc  = (v8f){0, 0, 0, 0, 0, 0, 0, 0};
      acc = wmma_bf16(a1_0, b0, acc);
      acc = wmma_bf16(a1_1, b1, acc);
      e[jt] = acc;
    }

    // online softmax of s = -E  (== softmax(rowmax(E) - E))
    float tm[8];
#pragma unroll
    for (int v = 0; v < 8; ++v) {
      float mm = -e[0][v];
      mm = fmaxf(mm, -e[1][v]);
      mm = fmaxf(mm, -e[2][v]);
      mm = fmaxf(mm, -e[3][v]);
      tm[v] = mm;
    }
#pragma unroll
    for (int off = 1; off < 16; off <<= 1) {
#pragma unroll
      for (int v = 0; v < 8; ++v)
        tm[v] = fmaxf(tm[v], __shfl_xor(tm[v], off, 16));
    }

    float scale[8], rsum[8];
#pragma unroll
    for (int v = 0; v < 8; ++v) {
      float mn = fmaxf(m_run[v], tm[v]);
      scale[v] = __expf(m_run[v] - mn);
      m_run[v] = mn;
      float s  = 0.0f;
#pragma unroll
      for (int jt = 0; jt < 4; ++jt) {
        float p = __expf(-e[jt][v] - mn);
        e[jt][v] = p;                       // reuse e[] to hold P
        s += p;
      }
      rsum[v] = s;
    }
#pragma unroll
    for (int off = 1; off < 16; off <<= 1) {
#pragma unroll
      for (int v = 0; v < 8; ++v)
        rsum[v] += __shfl_xor(rsum[v], off, 16);
    }
#pragma unroll
    for (int v = 0; v < 8; ++v) {
      l_run[v] = l_run[v] * scale[v] + rsum[v];
#pragma unroll
      for (int cb = 0; cb < 4; ++cb) o[cb][v] *= scale[v];
    }

    // write P tile to LDS as bf16 in [i][j] order (A2 layout source)
#pragma unroll
    for (int jt = 0; jt < 4; ++jt) {
#pragma unroll
      for (int v = 0; v < 8; ++v) {
        int rrow = v + 8 * hf;
        pls[wv][rrow * 64 + jt * 16 + n16] = f2bf(e[jt][v]);
      }
    }
    __syncthreads();

    // O[i, c] += P (16 x 64) @ f^T (64 x 16) for 4 c-blocks
    v16bf a2_0 = gatherA(pls[wv], 0, 0, 64, lane);
    v16bf a2_1 = gatherA(pls[wv], 0, 32, 64, lane);
#pragma unroll
    for (int cb = 0; cb < 4; ++cb) {
      v16bf b0 = gatherB(fBcj[cur], cb * 16, 0, 64, lane);
      v16bf b1 = gatherB(fBcj[cur], cb * 16, 32, 64, lane);
      o[cb] = wmma_bf16(a2_0, b0, o[cb]);
      o[cb] = wmma_bf16(a2_1, b1, o[cb]);
    }
    __syncthreads();   // fBcj[cur] reads done before tile n+2 DMA reuses it
  }

  // epilogue: out = gamma * (O / l) + feat
  const float g = gamma_p[0];
  float invl[8];
#pragma unroll
  for (int v = 0; v < 8; ++v) invl[v] = 1.0f / l_run[v];

#pragma unroll
  for (int cb = 0; cb < 4; ++cb) {
    int c = cb * 16 + n16;
#pragma unroll
    for (int v = 0; v < 8; ++v) {
      int i = i0 + wv * 16 + v + 8 * hf;
      long idx = ((long)(b * Cc + c) << 12) + i;
      out[idx] = g * (o[cb][v] * invl[v]) + feat[idx];
    }
  }
}

// ---------------------------------------------------------------------------
extern "C" void kernel_launch(void* const* d_in, const int* in_sizes, int n_in,
                              void* d_out, int out_size, void* d_ws,
                              size_t ws_size, hipStream_t stream) {
  (void)in_sizes; (void)n_in; (void)out_size; (void)ws_size;
  const float* x        = (const float*)d_in[0];
  const float* conv_w   = (const float*)d_in[1];
  const float* conv_b   = (const float*)d_in[2];
  const float* bn_scale = (const float*)d_in[3];
  const float* bn_bias  = (const float*)d_in[4];
  const float* prelu_a  = (const float*)d_in[5];
  const float* gamma    = (const float*)d_in[6];
  float* out = (float*)d_out;

  // workspace layout
  char* ws = (char*)d_ws;
  float*          y     = (float*)ws;                          // 4 MiB (reused as feat)
  unsigned short* fbf   = (unsigned short*)(ws + (4u << 20));  // 2 MiB
  float*          stats = (float*)(ws + (6u << 20));           // 128 f32
  float*          coef  = stats + 128;                         // 128 f32

  (void)hipMemsetAsync(stats, 0, 2 * Cc * sizeof(float), stream);

  const int total = Bb * Cc * Nn;                    // 1,048,576
  conv_bn_stats_kernel<<<total / 256, 256, 0, stream>>>(x, conv_w, conv_b, y,
                                                        stats);
  bn_finalize_kernel<<<1, 64, 0, stream>>>(stats, bn_scale, bn_bias, coef);
  bn_apply_prelu_kernel<<<total / 256, 256, 0, stream>>>(y, fbf, coef,
                                                         prelu_a);
  pam_attention_kernel<<<dim3(Nn / 64, Bb), 128, 0, stream>>>(fbf, y, gamma,
                                                              out);
}
